// Decoder_39599598469317
// MI455X (gfx1250) — matrix operand order repaired and checked by reading, not compiled, not run on hardware
//
#include <hip/hip_runtime.h>

// ---------------------------------------------------------------------------
// CDNA5 (gfx1250) transformer decoder forward.
// All GEMM-shaped work runs through v_wmma_f32_16x16x32_bf16 (wave32).
// ---------------------------------------------------------------------------

typedef __attribute__((ext_vector_type(16))) __bf16 v16bf;
typedef __attribute__((ext_vector_type(8)))  float  v8f;

#define DMODEL 512
#define NHEAD  8
#define DK     64
#define T1     1024
#define BATCH  8
#define NROWS  (BATCH * T1)   // 8192

struct U4x2 { uint4 a, b; };

static __device__ inline v16bf bc16(uint4 a, uint4 b) {
    U4x2 t{a, b};
    return __builtin_bit_cast(v16bf, t);
}

static __device__ inline v8f wmma_bf16(v16bf a, v16bf b, v8f c) {
    return __builtin_amdgcn_wmma_f32_16x16x32_bf16(
        false, a, false, b, (short)0, c, false, false);
}

static __device__ inline unsigned pack2bf(__bf16 a, __bf16 b) {
    return (unsigned)__builtin_bit_cast(unsigned short, a) |
           ((unsigned)__builtin_bit_cast(unsigned short, b) << 16);
}

static __device__ inline unsigned packf2(float a, float b) {
    return pack2bf((__bf16)a, (__bf16)b);
}

// ---- DPP16 butterfly reductions over a 16-lane half-wave (pure VALU) ------
template <int CTRL>
static __device__ inline float fdpp(float x) {
    return __builtin_bit_cast(float,
        __builtin_amdgcn_update_dpp(0, __builtin_bit_cast(int, x),
                                    CTRL, 0xF, 0xF, true));
}
static __device__ inline float rmax16(float x) {
    x = fmaxf(x, fdpp<0xB1>(x));    // quad_perm(1,0,3,2): xor 1
    x = fmaxf(x, fdpp<0x4E>(x));    // quad_perm(2,3,0,1): xor 2
    x = fmaxf(x, fdpp<0x141>(x));   // row_half_mirror: cross-quad
    x = fmaxf(x, fdpp<0x140>(x));   // row_mirror: cross-octet
    return x;
}
static __device__ inline float rsum16(float x) {
    x += fdpp<0xB1>(x);
    x += fdpp<0x4E>(x);
    x += fdpp<0x141>(x);
    x += fdpp<0x140>(x);
    return x;
}

// ---------------------------------------------------------------------------
// f32 -> bf16 conversion (grid-stride)
// ---------------------------------------------------------------------------
__global__ __launch_bounds__(256) void convert_f32_bf16(const float* __restrict__ in,
                                                        __bf16* __restrict__ out, int n) {
    int i = blockIdx.x * blockDim.x + threadIdx.x;
    int stride = gridDim.x * blockDim.x;
    for (; i < n; i += stride) out[i] = (__bf16)in[i];
}

// ---------------------------------------------------------------------------
// Embedding + sinusoid position + LayerNorm. One block per row (B*T1 rows).
// ---------------------------------------------------------------------------
__global__ __launch_bounds__(256) void embed_ln_kernel(const int* __restrict__ trg,
                                                       const float* __restrict__ emb,
                                                       const float* __restrict__ g,
                                                       const float* __restrict__ beta,
                                                       float* __restrict__ xo,
                                                       __bf16* __restrict__ xb) {
    const int r = blockIdx.x;          // row in [0, 8192)
    const int b = r >> 10;
    const int t = r & 1023;
    const int tid = threadIdx.x;

    float v[2];
#pragma unroll
    for (int e = 0; e < 2; ++e) {
        int i = tid + e * 256;
        float tok = 0.f;
        if (t > 0) {
            int token = trg[b * 1023 + (t - 1)];
            tok = emb[(size_t)token * DMODEL + i];
        }
        int ii = i & ~1;
        float freq = __powf(10000.f, -(float)ii / (float)DMODEL);
        float ang = (float)t * freq;
        float pos = (i & 1) ? __cosf(ang) : __sinf(ang);
        v[e] = tok + pos;
    }

    float s = v[0] + v[1];
    float q = v[0] * v[0] + v[1] * v[1];
#pragma unroll
    for (int off = 16; off >= 1; off >>= 1) {
        s += __shfl_xor(s, off, 32);
        q += __shfl_xor(q, off, 32);
    }
    __shared__ float ss[8], qq[8];
    __shared__ float mb, rb;
    int wave = tid >> 5, lane = tid & 31;
    if (lane == 0) { ss[wave] = s; qq[wave] = q; }
    __syncthreads();
    if (tid == 0) {
        float S = 0.f, Q = 0.f;
#pragma unroll
        for (int i = 0; i < 8; ++i) { S += ss[i]; Q += qq[i]; }
        float mean = S * (1.f / DMODEL);
        float var = Q * (1.f / DMODEL) - mean * mean;
        mb = mean;
        rb = rsqrtf(var + 1e-6f);
    }
    __syncthreads();
    float mean = mb, rstd = rb;
#pragma unroll
    for (int e = 0; e < 2; ++e) {
        int i = tid + e * 256;
        float o = (v[e] - mean) * rstd * g[i] + beta[i];
        xo[(size_t)r * DMODEL + i] = o;
        xb[(size_t)r * DMODEL + i] = (__bf16)o;
    }
}

// ---------------------------------------------------------------------------
// LayerNorm over pre-activation (residual already folded in by GEMM epilogue).
// ---------------------------------------------------------------------------
__global__ __launch_bounds__(256) void ln_kernel(const float* __restrict__ pre,
                                                 const float* __restrict__ g,
                                                 const float* __restrict__ beta,
                                                 float* __restrict__ xo,
                                                 __bf16* __restrict__ xb) {
    const int r = blockIdx.x;
    const int tid = threadIdx.x;
    const float* row = pre + (size_t)r * DMODEL;
    float v0 = row[tid], v1 = row[tid + 256];
    float s = v0 + v1, q = v0 * v0 + v1 * v1;
#pragma unroll
    for (int off = 16; off >= 1; off >>= 1) {
        s += __shfl_xor(s, off, 32);
        q += __shfl_xor(q, off, 32);
    }
    __shared__ float ss[8], qq[8];
    __shared__ float mb, rb;
    int wave = tid >> 5, lane = tid & 31;
    if (lane == 0) { ss[wave] = s; qq[wave] = q; }
    __syncthreads();
    if (tid == 0) {
        float S = 0.f, Q = 0.f;
#pragma unroll
        for (int i = 0; i < 8; ++i) { S += ss[i]; Q += qq[i]; }
        float mean = S * (1.f / DMODEL);
        float var = Q * (1.f / DMODEL) - mean * mean;
        mb = mean;
        rb = rsqrtf(var + 1e-6f);
    }
    __syncthreads();
    float mean = mb, rstd = rb;
    float o0 = (v0 - mean) * rstd * g[tid] + beta[tid];
    float o1 = (v1 - mean) * rstd * g[tid + 256] + beta[tid + 256];
    xo[(size_t)r * DMODEL + tid] = o0;
    xo[(size_t)r * DMODEL + tid + 256] = o1;
    xb[(size_t)r * DMODEL + tid] = (__bf16)o0;
    xb[(size_t)r * DMODEL + tid + 256] = (__bf16)o1;
}

// ---------------------------------------------------------------------------
// GEMM: C[M,N] = A_bf16[M,K] * B_f32[K,N] (+bias)(+res)(relu) -> bf16 or f32.
// Fusions are template parameters so the epilogue is branch-free.
// Block: 256 threads = 8 waves, 128x64 C tile, K-step 32.
// B is converted f32->bf16 while staging into LDS transposed [n][k]; each
// loader thread handles a k-pair so stores are packed ds_store_b32.
// ---------------------------------------------------------------------------
template <bool OUTF, bool BIAS, bool RES, bool RELU>
__global__ __launch_bounds__(256) void gemm_bf16_wmma(const __bf16* __restrict__ A,
                                                      const float* __restrict__ Bw,
                                                      const float* __restrict__ bias,
                                                      const float* __restrict__ res,
                                                      float* __restrict__ outF,
                                                      __bf16* __restrict__ outB,
                                                      int M, int N, int K) {
    __shared__ alignas(16) __bf16 As[128 * 40];   // [m][k], stride 40 (80B rows)
    __shared__ alignas(16) __bf16 Bst[64 * 40];   // [n][k], stride 40

    const int tid  = threadIdx.x;
    const int wave = tid >> 5;
    const int lane = tid & 31;
    const int hl   = lane >> 4;     // half-wave select
    const int n16  = lane & 15;
    const int bm = blockIdx.y * 128;
    const int bn = blockIdx.x * 64;

    // loader coordinates
    const int arow = tid >> 1, ac0 = (tid & 1) * 16;          // A: 1 row, 16 cols
    const int bkp  = tid & 15, bns = tid >> 4;                // B: k-pair, 4 n's

    v8f acc[4] = {};

    for (int k0 = 0; k0 < K; k0 += 32) {
        __syncthreads();
        // prefetch next K-step tiles into cache (global_prefetch_b8)
        if (k0 + 32 < K) {
            __builtin_prefetch(A  + (size_t)(bm + arow) * K + k0 + 32 + ac0, 0, 1);
            __builtin_prefetch(Bw + (size_t)(k0 + 32 + 2 * bkp) * N + bn + bns * 4, 0, 1);
        }
        // stage A: 128x32 bf16, coalesced 32B per thread
        {
            const uint4* src = (const uint4*)(A + (size_t)(bm + arow) * K + k0 + ac0);
            uint4 x0 = src[0], x1 = src[1];
            uint4* dst = (uint4*)(As + arow * 40 + ac0);
            dst[0] = x0; dst[1] = x1;
        }
        // stage B: 32x64 f32 -> bf16 transposed [n][k], packed b32 stores
        {
            const float4* s0 = (const float4*)(Bw + (size_t)(k0 + 2 * bkp) * N + bn + bns * 4);
            const float4* s1 = (const float4*)(Bw + (size_t)(k0 + 2 * bkp + 1) * N + bn + bns * 4);
            float4 f0 = s0[0], f1 = s1[0];
            float a0[4] = {f0.x, f0.y, f0.z, f0.w};
            float a1[4] = {f1.x, f1.y, f1.z, f1.w};
#pragma unroll
            for (int i = 0; i < 4; ++i)
                *(unsigned*)(Bst + (size_t)(bns * 4 + i) * 40 + 2 * bkp) = packf2(a0[i], a1[i]);
        }
        __syncthreads();

        // A fragment for this wave's 16-row slice
        const uint4* ap = (const uint4*)(As + (wave * 16 + n16) * 40);
        v16bf af = bc16(ap[2 * hl], ap[2 * hl + 1]);
#pragma unroll
        for (int nt = 0; nt < 4; ++nt) {
            const uint4* bp = (const uint4*)(Bst + (nt * 16 + n16) * 40 + hl * 16);
            v16bf bf = bc16(bp[0], bp[1]);
            acc[nt] = wmma_bf16(af, bf, acc[nt]);
        }
    }

    // branch-free epilogue
#pragma unroll
    for (int nt = 0; nt < 4; ++nt) {
        const int col = bn + nt * 16 + n16;
        float bv = 0.f;
        if constexpr (BIAS) bv = bias[col];
#pragma unroll
        for (int r = 0; r < 8; ++r) {
            const int row = bm + wave * 16 + r + 8 * hl;
            float v = acc[nt][r];
            if constexpr (BIAS) v += bv;
            if constexpr (RES)  v += res[(size_t)row * N + col];
            if constexpr (RELU) v = fmaxf(v, 0.f);
            if constexpr (OUTF) outF[(size_t)row * N + col] = v;
            else                outB[(size_t)row * N + col] = (__bf16)v;
        }
    }
}

// ---------------------------------------------------------------------------
// Flash attention (streaming softmax), bf16 WMMA.
// grid (16 qblocks, 8 heads, 8 batch), block 128 = 4 waves.
// Each wave owns a 16-row q slice of a 64-row q block; key tiles of 32.
// Row reductions use DPP16 (no LDS traffic); V staged transposed with
// packed ds_store_b32.
// ---------------------------------------------------------------------------
__global__ __launch_bounds__(128) void flash_attn_kernel(const __bf16* __restrict__ Q,
                                                         const __bf16* __restrict__ K,
                                                         const __bf16* __restrict__ V,
                                                         __bf16* __restrict__ O,
                                                         int causal) {
    const int qb = blockIdx.x;     // q block (64 rows)
    const int h  = blockIdx.y;
    const int b  = blockIdx.z;
    const int tid  = threadIdx.x;
    const int wave = tid >> 5;
    const int lane = tid & 31;
    const int hl   = lane >> 4;
    const int n16  = lane & 15;

    __shared__ alignas(16) __bf16 Kt[32 * 72];      // [key][d], stride 72
    __shared__ alignas(16) __bf16 Vt[64 * 40];      // [d][key], stride 40
    __shared__ alignas(16) __bf16 Ps[4 * 16 * 32];  // per-wave P tile [m][k]

    __bf16* Pw = Ps + wave * (16 * 32);

    const __bf16* qp0 = Q + (size_t)(b * T1) * DMODEL + h * DK;
    const __bf16* kp0 = K + (size_t)(b * T1) * DMODEL + h * DK;
    const __bf16* vp0 = V + (size_t)(b * T1) * DMODEL + h * DK;

    // Q fragments: rows qb*64 + wave*16 + (lane&15), dk split into two 32-chunks
    const int qrow = qb * 64 + wave * 16 + n16;
    const uint4* qv = (const uint4*)(qp0 + (size_t)qrow * DMODEL);
    v16bf aq0 = bc16(qv[2 * hl],     qv[2 * hl + 1]);
    v16bf aq1 = bc16(qv[4 + 2 * hl], qv[5 + 2 * hl]);

    v8f o[4] = {};
    float m_[8], l_[8];
#pragma unroll
    for (int r = 0; r < 8; ++r) { m_[r] = -1e30f; l_[r] = 0.f; }

    const int ntiles = causal ? (2 * qb + 2) : (T1 / 32);
    const float sc = 0.125f;   // 1/sqrt(64)

    // loader coordinates
    const int krow = tid >> 2, kseg = tid & 3;   // K: 1 key, 16 d's
    const int vkp  = tid & 15, vds  = tid >> 4;  // V: key-pair, 8 d's

    for (int t = 0; t < ntiles; ++t) {
        const int kt = t * 32;
        __syncthreads();
        // cooperative stage: Kt row-major (b128 stores), Vt transposed (b32)
        {
            const uint4* ks = (const uint4*)(kp0 + (size_t)(kt + krow) * DMODEL + kseg * 16);
            uint4 a = ks[0], bq = ks[1];
            uint4* kd = (uint4*)(Kt + krow * 72 + kseg * 16);
            kd[0] = a; kd[1] = bq;

            const uint4* vs0 = (const uint4*)(vp0 + (size_t)(kt + 2 * vkp) * DMODEL + vds * 8);
            const uint4* vs1 = (const uint4*)(vp0 + (size_t)(kt + 2 * vkp + 1) * DMODEL + vds * 8);
            uint4 va = vs0[0], vb = vs1[0];
            __bf16 t0[8], t1[8];
            *(uint4*)t0 = va;
            *(uint4*)t1 = vb;
#pragma unroll
            for (int i = 0; i < 8; ++i)
                *(unsigned*)(Vt + (size_t)(vds * 8 + i) * 40 + 2 * vkp) = pack2bf(t0[i], t1[i]);
        }
        __syncthreads();

        // S = Q K^T : two 16-col blocks, each K=64 via two WMMAs
        v8f zero = {};
        v8f s0, s1;
        {
            const uint4* kr = (const uint4*)(Kt + (0 * 16 + n16) * 72);
            v16bf b0 = bc16(kr[2 * hl],     kr[2 * hl + 1]);
            v16bf b1 = bc16(kr[4 + 2 * hl], kr[4 + 2 * hl + 1]);
            s0 = wmma_bf16(aq0, b0, zero);
            s0 = wmma_bf16(aq1, b1, s0);
        }
        {
            const uint4* kr = (const uint4*)(Kt + (1 * 16 + n16) * 72);
            v16bf b0 = bc16(kr[2 * hl],     kr[2 * hl + 1]);
            v16bf b1 = bc16(kr[4 + 2 * hl], kr[4 + 2 * hl + 1]);
            s1 = wmma_bf16(aq0, b0, zero);
            s1 = wmma_bf16(aq1, b1, s1);
        }

        // online softmax per row (rows live across 16 lanes of each half-wave)
#pragma unroll
        for (int r = 0; r < 8; ++r) {
            float v0 = s0[r] * sc;
            float v1 = s1[r] * sc;
            if (causal) {
                int qg = qb * 64 + wave * 16 + r + 8 * hl;
                if (kt + n16 > qg)      v0 = -1e30f;
                if (kt + 16 + n16 > qg) v1 = -1e30f;
            }
            float mx = rmax16(fmaxf(v0, v1));
            float mn = fmaxf(m_[r], mx);
            float alpha = __expf(m_[r] - mn);
            float e0 = __expf(v0 - mn);
            float e1 = __expf(v1 - mn);
            float rs = rsum16(e0 + e1);
            l_[r] = l_[r] * alpha + rs;
            m_[r] = mn;
            o[0][r] *= alpha; o[1][r] *= alpha; o[2][r] *= alpha; o[3][r] *= alpha;
            // C-layout -> LDS row-major P tile (bf16)
            int mrow = r + 8 * hl;
            Pw[mrow * 32 + n16]      = (__bf16)e0;
            Pw[mrow * 32 + 16 + n16] = (__bf16)e1;
        }
        asm volatile("s_wait_dscnt 0" ::: "memory");

        // P as A-fragment (16x32)
        const uint4* pp = (const uint4*)(Pw + n16 * 32);
        v16bf ap = bc16(pp[2 * hl], pp[2 * hl + 1]);

        // O += P * V  (four 16-col d blocks)
#pragma unroll
        for (int nt = 0; nt < 4; ++nt) {
            const uint4* vv = (const uint4*)(Vt + (nt * 16 + n16) * 40 + hl * 16);
            v16bf bv = bc16(vv[0], vv[1]);
            o[nt] = wmma_bf16(ap, bv, o[nt]);
        }
    }

    // epilogue: divide by row sums, store bf16
#pragma unroll
    for (int nt = 0; nt < 4; ++nt) {
#pragma unroll
        for (int r = 0; r < 8; ++r) {
            int qg = qb * 64 + wave * 16 + r + 8 * hl;
            float val = o[nt][r] / l_[r];
            O[(size_t)(b * T1 + qg) * DMODEL + h * DK + nt * 16 + n16] = (__bf16)val;
        }
    }
}

// ---------------------------------------------------------------------------
// Host launcher
// ---------------------------------------------------------------------------
extern "C" void kernel_launch(void* const* d_in, const int* in_sizes, int n_in,
                              void* d_out, int out_size, void* d_ws, size_t ws_size,
                              hipStream_t stream) {
    (void)in_sizes; (void)n_in; (void)out_size; (void)ws_size;

    const int*   trg_seq = (const int*)d_in[0];
    const float* enc     = (const float*)d_in[2];
    const float* emb     = (const float*)d_in[4];
    const float* ln0_g   = (const float*)d_in[5];
    const float* ln0_b   = (const float*)d_in[6];
    const float* wq_s = (const float*)d_in[7];
    const float* wk_s = (const float*)d_in[8];
    const float* wv_s = (const float*)d_in[9];
    const float* wo_s = (const float*)d_in[10];
    const float* wq_c = (const float*)d_in[11];
    const float* wk_c = (const float*)d_in[12];
    const float* wv_c = (const float*)d_in[13];
    const float* wo_c = (const float*)d_in[14];
    const float* ln1_g = (const float*)d_in[15];
    const float* ln2_g = (const float*)d_in[16];
    const float* ln3_g = (const float*)d_in[17];
    const float* ln1_b = (const float*)d_in[18];
    const float* ln2_b = (const float*)d_in[19];
    const float* ln3_b = (const float*)d_in[20];
    const float* w1 = (const float*)d_in[21];
    const float* b1 = (const float*)d_in[22];
    const float* w2 = (const float*)d_in[23];
    const float* b2 = (const float*)d_in[24];

    // workspace partitions
    char* p = (char*)d_ws;
    auto alloc = [&](size_t bytes) -> void* {
        void* r = (void*)p;
        p += (bytes + 255) & ~(size_t)255;
        return r;
    };
    const size_t actE = (size_t)NROWS * DMODEL;            // 8192*512
    __bf16* ENCB = (__bf16*)alloc(actE * 2);
    float*  X    = (float*) alloc(actE * 4);
    __bf16* XB   = (__bf16*)alloc(actE * 2);
    __bf16* QB   = (__bf16*)alloc(actE * 2);
    __bf16* KB   = (__bf16*)alloc(actE * 2);
    __bf16* VB   = (__bf16*)alloc(actE * 2);
    __bf16* ATTN = (__bf16*)alloc(actE * 2);
    float*  PRE  = (float*) alloc(actE * 4);
    __bf16* HB   = (__bf16*)alloc((size_t)NROWS * 2048 * 2);

    // encoder output -> bf16
    convert_f32_bf16<<<4096, 256, 0, stream>>>(enc, ENCB, (int)actE);

    // embedding + position + ln0
    embed_ln_kernel<<<NROWS, 256, 0, stream>>>(trg_seq, emb, ln0_g, ln0_b, X, XB);

    const dim3 gP(DMODEL / 64, NROWS / 128);      // projections: N=512
    const dim3 gF1(2048 / 64, NROWS / 128);       // FFN1: N=2048
    const dim3 gAtt(16, NHEAD, BATCH);

    for (int l = 0; l < 6; ++l) {
        const size_t wo512  = (size_t)l * DMODEL * DMODEL;
        const size_t woff1  = (size_t)l * DMODEL * 2048;
        const size_t woff2  = (size_t)l * 2048 * DMODEL;

        // ---- self attention ----
        gemm_bf16_wmma<false, false, false, false><<<gP, 256, 0, stream>>>(
            XB, wq_s + wo512, nullptr, nullptr, nullptr, QB, NROWS, DMODEL, DMODEL);
        gemm_bf16_wmma<false, false, false, false><<<gP, 256, 0, stream>>>(
            XB, wk_s + wo512, nullptr, nullptr, nullptr, KB, NROWS, DMODEL, DMODEL);
        gemm_bf16_wmma<false, false, false, false><<<gP, 256, 0, stream>>>(
            XB, wv_s + wo512, nullptr, nullptr, nullptr, VB, NROWS, DMODEL, DMODEL);
        flash_attn_kernel<<<gAtt, 128, 0, stream>>>(QB, KB, VB, ATTN, 1);
        gemm_bf16_wmma<true, false, true, false><<<gP, 256, 0, stream>>>(
            ATTN, wo_s + wo512, nullptr, X, PRE, nullptr, NROWS, DMODEL, DMODEL);
        ln_kernel<<<NROWS, 256, 0, stream>>>(PRE, ln1_g + l * DMODEL, ln1_b + l * DMODEL, X, XB);

        // ---- cross attention ----
        gemm_bf16_wmma<false, false, false, false><<<gP, 256, 0, stream>>>(
            XB, wq_c + wo512, nullptr, nullptr, nullptr, QB, NROWS, DMODEL, DMODEL);
        gemm_bf16_wmma<false, false, false, false><<<gP, 256, 0, stream>>>(
            ENCB, wk_c + wo512, nullptr, nullptr, nullptr, KB, NROWS, DMODEL, DMODEL);
        gemm_bf16_wmma<false, false, false, false><<<gP, 256, 0, stream>>>(
            ENCB, wv_c + wo512, nullptr, nullptr, nullptr, VB, NROWS, DMODEL, DMODEL);
        flash_attn_kernel<<<gAtt, 128, 0, stream>>>(QB, KB, VB, ATTN, 0);
        gemm_bf16_wmma<true, false, true, false><<<gP, 256, 0, stream>>>(
            ATTN, wo_c + wo512, nullptr, X, PRE, nullptr, NROWS, DMODEL, DMODEL);
        ln_kernel<<<NROWS, 256, 0, stream>>>(PRE, ln2_g + l * DMODEL, ln2_b + l * DMODEL, X, XB);

        // ---- FFN ----
        gemm_bf16_wmma<false, true, false, true><<<gF1, 256, 0, stream>>>(
            XB, w1 + woff1, b1 + (size_t)l * 2048, nullptr, nullptr, HB, NROWS, 2048, DMODEL);
        gemm_bf16_wmma<true, true, true, false><<<gP, 256, 0, stream>>>(
            HB, w2 + woff2, b2 + (size_t)l * DMODEL, X, PRE, nullptr, NROWS, DMODEL, 2048);
        ln_kernel<<<NROWS, 256, 0, stream>>>(PRE, ln3_g + l * DMODEL, ln3_b + l * DMODEL, X, XB);
    }

    hipMemcpyAsync(d_out, X, actE * sizeof(float), hipMemcpyDeviceToDevice, stream);
}
